// TransformerClassifier_34668976013327
// MI455X (gfx1250) — compile-verified, matrix operand order checked by reading, not run on hardware
//
#include <hip/hip_runtime.h>

#define DM   1024           // d_model
#define SEQ  2048           // sequence length
#define NB   4              // batch
#define MROWS (NB * SEQ)    // 8192 flattened rows
#define QT   32             // query rows per attention workgroup

typedef __attribute__((ext_vector_type(16))) __bf16 v16bf;
typedef __attribute__((ext_vector_type(8)))  __bf16 v8bf;
typedef __attribute__((ext_vector_type(8)))  float  v8f;

#if defined(__has_builtin)
#if __has_builtin(__builtin_amdgcn_tensor_load_to_lds) && \
    __has_builtin(__builtin_amdgcn_s_wait_tensorcnt)
#define HAVE_TDM 1
#endif
#endif

#ifdef HAVE_TDM
typedef __attribute__((ext_vector_type(4))) unsigned int u32x4;
typedef __attribute__((ext_vector_type(4))) int          i32x4;
typedef __attribute__((ext_vector_type(8))) int          i32x8;

// 2-D tile DMA: global (tile_rows x DM bf16, row stride DM) -> LDS,
// descriptor packed per ISA 8.3/8.4 (count=1, type=2, data_size=2B).
static __device__ __forceinline__ void tdm_load_tile_2d(
    unsigned lds_byte_addr, const void* gptr, unsigned tile_rows)
{
    unsigned long long ga = (unsigned long long)(uintptr_t)gptr;
    u32x4 g0;
    g0[0] = 1u;                                           // count=1, user desc
    g0[1] = lds_byte_addr;                                // lds_addr
    g0[2] = (unsigned)(ga & 0xFFFFFFFFu);                 // global_addr[31:0]
    g0[3] = (unsigned)((ga >> 32) & 0x01FFFFFFu)          // global_addr[56:32]
          | (2u << 30);                                   // type=2 ("image")
    i32x8 g1;
    g1[0] = 0x00010000;                                   // data_size=1 (2B)
    g1[1] = (int)(((unsigned)DM & 0xFFFFu) << 16);        // tensor_dim0[15:0]
    g1[2] = (int)(((unsigned)SEQ & 0xFFFFu) << 16);       // dim0 hi=0 | tensor_dim1 lo
    g1[3] = (int)(((unsigned)DM & 0xFFFFu) << 16);        // dim1 hi=0 | tile_dim0
    g1[4] = (int)tile_rows;                               // tile_dim1 (tile_dim2=0)
    g1[5] = DM;                                           // tensor_dim0_stride[31:0]
    g1[6] = 0;                                            // stride hi | dim1_stride lo
    g1[7] = 0;
    i32x4 z4; z4[0] = z4[1] = z4[2] = z4[3] = 0;
#if __clang_major__ >= 23
    i32x8 z8;
#pragma unroll
    for (int i = 0; i < 8; ++i) z8[i] = 0;
    __builtin_amdgcn_tensor_load_to_lds(g0, g1, z4, z4, z8, 0);
#else
    __builtin_amdgcn_tensor_load_to_lds(g0, g1, z4, z4, 0);
#endif
}
#endif  // HAVE_TDM

// ---------------------------------------------------------------------------
// Per-lane loader for a 16x32 bf16 WMMA operand tile from row-major storage.
// ISA 7.12.2: lane = {row = lane&15, K-half = lane>>4}; VGPR0..3 hold
// K = half*8+0..7, VGPR4..7 hold K = 16+half*8+0..7 -> two 16B loads/lane.
// ---------------------------------------------------------------------------
static __device__ __forceinline__ v16bf load_tile16x32(const __bf16* base, int ld) {
    const int lane = threadIdx.x & 31;
    const int r = lane & 15;
    const int h = (lane >> 4) & 1;
    const v8bf lo = *reinterpret_cast<const v8bf*>(base + (size_t)r * ld + h * 8);
    const v8bf hi = *reinterpret_cast<const v8bf*>(base + (size_t)r * ld + 16 + h * 8);
    return __builtin_shufflevector(lo, hi, 0,1,2,3,4,5,6,7,8,9,10,11,12,13,14,15);
}

static __device__ __forceinline__ v8f vzero8() {
    v8f z;
#pragma unroll
    for (int i = 0; i < 8; ++i) z[i] = 0.0f;
    return z;
}

static __device__ __forceinline__ v8f wmma_bf16(v16bf a, v16bf b, v8f c) {
    return __builtin_amdgcn_wmma_f32_16x16x32_bf16(
        false, a, false, b, (short)0, c, false, false);
}

static __device__ __forceinline__ float wave_max32(float v) {
#pragma unroll
    for (int m = 16; m >= 1; m >>= 1) v = fmaxf(v, __shfl_xor(v, m, 32));
    return v;
}
static __device__ __forceinline__ float wave_sum32(float v) {
#pragma unroll
    for (int m = 16; m >= 1; m >>= 1) v += __shfl_xor(v, m, 32);
    return v;
}

// ---------------------------------------------------------------------------
__global__ void cvt_f2bf(const float* __restrict__ in, __bf16* __restrict__ out, int n) {
    int i = blockIdx.x * blockDim.x + threadIdx.x;
    int stride = gridDim.x * blockDim.x;
    for (; i < n; i += stride) out[i] = (__bf16)in[i];
}

// fp32 DMxDM -> bf16 transposed so the WMMA B operand reads contiguously.
__global__ void cvt_f2bf_T(const float* __restrict__ in, __bf16* __restrict__ out) {
    int k = blockIdx.x * 16 + threadIdx.x;
    int n = blockIdx.y * 16 + threadIdx.y;
    out[(size_t)n * DM + k] = (__bf16)in[(size_t)k * DM + n];
}

// ---------------------------------------------------------------------------
// GEMM: C[MROWS x DM] = A * W, with W pre-transposed (Bt[ncol][k]).
// Wave tile 32x64 (2 A tiles reuse 4 shared B tiles -> 6 loads / 8 WMMA).
// WG = 8 waves arranged 2(M) x 4(N): WG tile 64 x 256.
// mode 0: row-major bf16 C.  mode 1: per-batch transposed store (Vt).
// ---------------------------------------------------------------------------
__global__ __launch_bounds__(256) void gemm_bf16_qkv(
    const __bf16* __restrict__ A,
    const __bf16* __restrict__ Bt,
    __bf16* __restrict__ C,
    int mode)
{
    const int wave = threadIdx.x >> 5;
    const int lane = threadIdx.x & 31;
    const int rowbase = blockIdx.x * 64  + (wave & 1) * 32;
    const int colbase = blockIdx.y * 256 + (wave >> 1) * 64;

    v8f acc[2][4];
#pragma unroll
    for (int rh = 0; rh < 2; ++rh)
#pragma unroll
        for (int t = 0; t < 4; ++t) acc[rh][t] = vzero8();

    for (int kc = 0; kc < DM / 32; ++kc) {
        const __bf16* abase = A + (size_t)rowbase * DM + kc * 32;
        if (kc + 1 < DM / 32)
            __builtin_prefetch(abase + 32, 0, 1);   // global_prefetch_b8
        v16bf b[4];
#pragma unroll
        for (int t = 0; t < 4; ++t)
            b[t] = load_tile16x32(Bt + (size_t)(colbase + t * 16) * DM + kc * 32, DM);
        v16bf a0 = load_tile16x32(abase, DM);
        v16bf a1 = load_tile16x32(abase + (size_t)16 * DM, DM);
#pragma unroll
        for (int t = 0; t < 4; ++t) {
            acc[0][t] = wmma_bf16(a0, b[t], acc[0][t]);
            acc[1][t] = wmma_bf16(a1, b[t], acc[1][t]);
        }
    }

    const int h = lane >> 4, c = lane & 15;
#pragma unroll
    for (int rh = 0; rh < 2; ++rh)
#pragma unroll
        for (int t = 0; t < 4; ++t)
#pragma unroll
            for (int v = 0; v < 8; ++v) {
                int row = rowbase + rh * 16 + v + 8 * h;
                int col = colbase + t * 16 + c;
                if (mode == 0) {
                    C[(size_t)row * DM + col] = (__bf16)acc[rh][t][v];
                } else {
                    int n = row >> 11;           // / SEQ
                    int l = row & (SEQ - 1);
                    C[((size_t)n * DM + col) * SEQ + l] = (__bf16)acc[rh][t][v];
                }
            }
}

// ---------------------------------------------------------------------------
// Attention: one WG per (32-query-row tile, batch).
// Phase 1: S = Q*K^T/sqrt(D) -> bf16 in LDS; each K tile feeds 2 WMMAs.
// Phase 2: per-row softmax in-wave; exp written back in place (P, bf16);
//          1/sum deferred to the epilogue.
// Phase 3: O = P * V via pre-transposed Vt; each V tile feeds 2 WMMAs.
// LDS: 64KB Q + 128KB S/P (within the 320KB WGP pool).
// ---------------------------------------------------------------------------
__global__ __launch_bounds__(256) void attn_bf16(
    const __bf16* __restrict__ Q,   // (NB*SEQ) x DM
    const __bf16* __restrict__ K,   // (NB*SEQ) x DM
    const __bf16* __restrict__ Vt,  // NB x DM x SEQ
    float* __restrict__ O)          // NB x SEQ x DM
{
    __shared__ __align__(16) __bf16 Qs[QT * DM];    // 64 KB
    __shared__ __align__(16) __bf16 Ss[QT * SEQ];   // 128 KB
    __shared__ float rowscale[QT];

    const int n = blockIdx.y;
    const int qbase = blockIdx.x * QT;
    const int tid = threadIdx.x;
    const int wave = tid >> 5, lane = tid & 31;

    // ---- Stage the QT x DM Q tile into LDS (TDM if available).
    const __bf16* qsrc = Q + ((size_t)(n * SEQ + qbase)) * DM;
#ifdef HAVE_TDM
    if (wave == 0) {
        tdm_load_tile_2d((unsigned)(uintptr_t)Qs, qsrc, QT);
        __builtin_amdgcn_s_wait_tensorcnt(0);
    }
#else
    {
        const uint4* src = reinterpret_cast<const uint4*>(qsrc);
        uint4* dst = reinterpret_cast<uint4*>(Qs);
#pragma unroll
        for (int i = 0; i < (QT * DM * 2 / 16) / 256; ++i)
            dst[tid + i * 256] = src[tid + i * 256];
    }
#endif
    __syncthreads();

    const float scale = 0.03125f;  // 1/sqrt(1024)

    // ---- Phase 1: 128 key tiles, 16 per wave; 2 query halves per K tile.
    for (int kt = wave * 16; kt < wave * 16 + 16; ++kt) {
        const int keybase = kt * 16;
        v8f acc0 = vzero8(), acc1 = vzero8();
        for (int kc = 0; kc < DM / 32; ++kc) {
            v16bf b  = load_tile16x32(K + ((size_t)(n * SEQ + keybase)) * DM + kc * 32, DM);
            v16bf a0 = load_tile16x32(Qs + kc * 32, DM);
            v16bf a1 = load_tile16x32(Qs + (size_t)16 * DM + kc * 32, DM);
            acc0 = wmma_bf16(a0, b, acc0);
            acc1 = wmma_bf16(a1, b, acc1);
        }
        const int h = lane >> 4, c = lane & 15;
#pragma unroll
        for (int v = 0; v < 8; ++v) {
            int r = v + 8 * h;
            Ss[(size_t)r * SEQ + keybase + c]        = (__bf16)(acc0[v] * scale);
            Ss[(size_t)(r + 16) * SEQ + keybase + c] = (__bf16)(acc1[v] * scale);
        }
    }
    __syncthreads();

    // ---- Phase 2: softmax, 4 rows per wave (whole row per wave).
    for (int r = wave * 4; r < wave * 4 + 4; ++r) {
        float m = -3.0e38f;
        for (int j = lane; j < SEQ; j += 32) m = fmaxf(m, (float)Ss[(size_t)r * SEQ + j]);
        m = wave_max32(m);
        float s = 0.0f;
        for (int j = lane; j < SEQ; j += 32) {
            float e = __expf((float)Ss[(size_t)r * SEQ + j] - m);
            s += e;
            Ss[(size_t)r * SEQ + j] = (__bf16)e;   // same slot, same lane
        }
        s = wave_sum32(s);
        if (lane == 0) rowscale[r] = 1.0f / s;
    }
    __syncthreads();

    // ---- Phase 3: O = P * V. Wave owns D columns [wave*128, +128).
    const int colbase = wave * 128;
    v8f acc[2][8];
#pragma unroll
    for (int rh = 0; rh < 2; ++rh)
#pragma unroll
        for (int t = 0; t < 8; ++t) acc[rh][t] = vzero8();

    for (int kc = 0; kc < SEQ / 32; ++kc) {
        v16bf a0 = load_tile16x32(Ss + kc * 32, SEQ);
        v16bf a1 = load_tile16x32(Ss + (size_t)16 * SEQ + kc * 32, SEQ);
#pragma unroll
        for (int t = 0; t < 8; ++t) {
            v16bf b = load_tile16x32(
                Vt + ((size_t)n * DM + colbase + t * 16) * SEQ + kc * 32, SEQ);
            acc[0][t] = wmma_bf16(a0, b, acc[0][t]);
            acc[1][t] = wmma_bf16(a1, b, acc[1][t]);
        }
    }

    const int h = lane >> 4, c = lane & 15;
#pragma unroll
    for (int rh = 0; rh < 2; ++rh)
#pragma unroll
        for (int t = 0; t < 8; ++t)
#pragma unroll
            for (int v = 0; v < 8; ++v) {
                int r = rh * 16 + v + 8 * h;
                O[((size_t)(n * SEQ + qbase + r)) * DM + colbase + t * 16 + c] =
                    acc[rh][t][v] * rowscale[r];
            }
}

// ---------------------------------------------------------------------------
__global__ __launch_bounds__(256) void pool_head(
    const float* __restrict__ H, const float* __restrict__ WO,
    const float* __restrict__ bO, float* __restrict__ out)
{
    __shared__ float red[256];
    const int n = blockIdx.x;
    float acc = 0.0f;
    for (int idx = threadIdx.x; idx < SEQ * DM; idx += 256) {
        int d = idx & (DM - 1);
        acc += H[(size_t)n * SEQ * DM + idx] * WO[d];
    }
    red[threadIdx.x] = acc;
    __syncthreads();
    for (int s = 128; s > 0; s >>= 1) {
        if (threadIdx.x < s) red[threadIdx.x] += red[threadIdx.x + s];
        __syncthreads();
    }
    if (threadIdx.x == 0) {
        float logit = red[0] * (1.0f / (float)SEQ) + bO[0];
        out[n] = 1.0f / (1.0f + __expf(-logit));
    }
}

// ---------------------------------------------------------------------------
extern "C" void kernel_launch(void* const* d_in, const int* in_sizes, int n_in,
                              void* d_out, int out_size, void* d_ws, size_t ws_size,
                              hipStream_t stream) {
    const float* X  = (const float*)d_in[0];
    const float* WO = (const float*)d_in[7];
    const float* bO = (const float*)d_in[8];

    char* ws = (char*)d_ws;
    size_t off = 0;
    auto take = [&](size_t bytes) -> void* {
        void* p = ws + off;
        off += (bytes + 255) & ~(size_t)255;
        return p;
    };
    __bf16* Xh = (__bf16*)take((size_t)MROWS * DM * 2);
    __bf16* Wt = (__bf16*)take((size_t)6 * DM * DM * 2);
    __bf16* Qb = (__bf16*)take((size_t)MROWS * DM * 2);
    __bf16* Kb = (__bf16*)take((size_t)MROWS * DM * 2);
    __bf16* Vt = (__bf16*)take((size_t)MROWS * DM * 2);
    float*  Hf = (float*)take((size_t)MROWS * DM * 4);
    if (off > ws_size) return;

    for (int w = 0; w < 6; ++w)
        cvt_f2bf_T<<<dim3(DM / 16, DM / 16), dim3(16, 16), 0, stream>>>(
            (const float*)d_in[1 + w], Wt + (size_t)w * DM * DM);
    cvt_f2bf<<<2048, 256, 0, stream>>>(X, Xh, MROWS * DM);

    for (int layer = 0; layer < 2; ++layer) {
        const __bf16* wq = Wt + (size_t)(layer * 3 + 0) * DM * DM;
        const __bf16* wk = Wt + (size_t)(layer * 3 + 1) * DM * DM;
        const __bf16* wv = Wt + (size_t)(layer * 3 + 2) * DM * DM;

        gemm_bf16_qkv<<<dim3(MROWS / 64, DM / 256), 256, 0, stream>>>(Xh, wq, Qb, 0);
        gemm_bf16_qkv<<<dim3(MROWS / 64, DM / 256), 256, 0, stream>>>(Xh, wk, Kb, 0);
        gemm_bf16_qkv<<<dim3(MROWS / 64, DM / 256), 256, 0, stream>>>(Xh, wv, Vt, 1);

        attn_bf16<<<dim3(SEQ / QT, NB), 256, 0, stream>>>(Qb, Kb, Vt, Hf);

        if (layer == 0)
            cvt_f2bf<<<2048, 256, 0, stream>>>(Hf, Xh, MROWS * DM);
    }

    pool_head<<<NB, 256, 0, stream>>>(Hf, WO, bO, (float*)d_out);
}
